// StructuredSSMCore_10453950399011
// MI455X (gfx1250) — matrix operand order repaired and checked by reading, not compiled, not run on hardware
//
#include <hip/hip_runtime.h>

// ---------------------------------------------------------------------------
// StructuredSSMCore for MI455X (gfx1250, wave32, WMMA)
//   y = scan(x = x*a_bar + u_tilde@B_bar^T) @ C^T + u_tilde @ D^T
// Phases:
//   0) build fused bf16 hi/lo weight matrix W = [b_int*B ; D]  (320 x 1024)
//      + bf16 hi/lo C, + pow table a^(t+1)
//   1) big GEMM  u_tilde(16384x1024) @ W^T -> Bu(...x64) and Dy(...x256)
//      using v_wmma_f32_16x16x32_bf16 with bf16x3 error-compensated products
//   2) chunked local scans (L=128, 8192 parallel chunk tasks)
//   3) tiny sequential carry pass over 16 chunks per (b,n)
//   4) output GEMM (xs_local + a^(t+1)*carry) @ C^T + Dy -> y, fused fixup
// ---------------------------------------------------------------------------

typedef __attribute__((ext_vector_type(16))) __bf16 bf16x16;
typedef __attribute__((ext_vector_type(8)))  float  f32x8;
typedef __attribute__((ext_vector_type(4)))  float  f32x4;

#define WMMA_BF16(A, B, C) \
  __builtin_amdgcn_wmma_f32_16x16x32_bf16(false, (A), false, (B), (short)0, (C), false, false)

constexpr int kB  = 8;
constexpr int kS  = 2048;
constexpr int kDs = 512;          // D_STREAM
constexpr int kDd = 1024;         // 2*D_STREAM (fused K)
constexpr int kDa = 256;          // D_ACT
constexpr int kN  = 64;           // N_STATE
constexpr int kM  = kB * kS;      // 16384 rows
constexpr int kNW = kN + kDa;     // 320 fused weight rows
constexpr int kL  = 128;          // scan chunk length
constexpr int kNC = kS / kL;      // 16 chunks per sequence

__device__ __forceinline__ void bsplit(float x, __bf16& h, __bf16& l) {
  h = (__bf16)x;                  // hi part
  l = (__bf16)(x - (float)h);     // residual (exact subtraction)
}

// ---------------- Phase 0: weight preparation -------------------------------

__global__ void ssm_prep_w(const float* __restrict__ lnl, const float* __restrict__ logdelta,
                           const float* __restrict__ Be, const float* __restrict__ Bf,
                           const float* __restrict__ De, const float* __restrict__ Df,
                           __bf16* __restrict__ Wh, __bf16* __restrict__ Wl) {
  int idx = blockIdx.x * blockDim.x + threadIdx.x;   // < 320*1024
  int n = idx >> 10;
  int k = idx & 1023;
  float w;
  if (n < kN) {
    float delta = expf(logdelta[0]);
    float lam   = -expf(lnl[n]);
    float abar  = expf(delta * lam);
    float bint  = (abar - 1.0f) / lam;
    float bval  = (k < kDs) ? Be[n * kDs + k] : Bf[n * kDs + (k - kDs)];
    w = bint * bval;
  } else {
    int a = n - kN;
    w = (k < kDs) ? De[a * kDs + k] : Df[a * kDs + (k - kDs)];
  }
  __bf16 h, l;
  bsplit(w, h, l);
  Wh[idx] = h;
  Wl[idx] = l;
}

__global__ void ssm_prep_c(const float* __restrict__ C,
                           __bf16* __restrict__ Ch, __bf16* __restrict__ Cl) {
  int idx = blockIdx.x * blockDim.x + threadIdx.x;   // < 256*64
  __bf16 h, l;
  bsplit(C[idx], h, l);
  Ch[idx] = h;
  Cl[idx] = l;
}

__global__ void ssm_prep_pow(const float* __restrict__ lnl, const float* __restrict__ logdelta,
                             float* __restrict__ powt) {
  int idx = blockIdx.x * blockDim.x + threadIdx.x;   // < 128*64
  int t = idx >> 6, n = idx & 63;
  float lnA = -expf(lnl[n]) * expf(logdelta[0]);     // delta*lambda (negative)
  powt[idx] = expf(lnA * (float)(t + 1));            // a_bar^(t+1)
}

// ---------------- Phase 1: fused input GEMM ---------------------------------
// One wave per (16-row M tile, group of 5 N tiles). K = 1024, 32 k-steps.
// Per k-step: 4x b128 f32 loads for A, split to bf16 hi/lo; per n-tile 2+2
// b128 bf16 loads for W hi/lo; 3 WMMAs (bf16x3 compensated product).

__global__ __launch_bounds__(32) void ssm_gemm_in(const float* __restrict__ u,
                                                  const float* __restrict__ uref,
                                                  const __bf16* __restrict__ Wh,
                                                  const __bf16* __restrict__ Wl,
                                                  float* __restrict__ Bu,
                                                  float* __restrict__ Dy) {
  const int mtile = blockIdx.x;          // 0..1023
  const int ngrp  = blockIdx.y;          // 0..3 (5 n-tiles each)
  const int l = threadIdx.x;
  const int h = l >> 4;                  // lane half selects K sub-group
  const int m = l & 15;                  // A-matrix row / C-matrix N lane
  const size_t rowA = (size_t)mtile * 16 + m;

  f32x8 acc[5];
  #pragma unroll
  for (int j = 0; j < 5; ++j) acc[j] = (f32x8)(0.0f);

  for (int k0 = 0; k0 < kDd; k0 += 32) {
    const float* src = (k0 < kDs) ? u : uref;
    const size_t base = rowA * kDs + (size_t)(k0 & (kDs - 1));
    // prefetch next k window (speculative, dropped on translation miss)
    __builtin_prefetch(src + base + 8 * h + 32, 0, 0);

    f32x4 av[4];
    av[0] = *(const f32x4*)(src + base + 16 * h);
    av[1] = *(const f32x4*)(src + base + 16 * h + 4);
    av[2] = *(const f32x4*)(src + base + 16 * h + 8);
    av[3] = *(const f32x4*)(src + base + 16 * h + 12);

    bf16x16 ah, al;
    #pragma unroll
    for (int i = 0; i < 16; ++i) {
      __bf16 hi, lo;
      bsplit(av[i >> 2][i & 3], hi, lo);
      ah[i] = hi;
      al[i] = lo;
    }

    #pragma unroll
    for (int j = 0; j < 5; ++j) {
      const int wrow = (ngrp * 5 + j) * 16 + m;      // B-matrix N = lane&15
      const size_t woff = (size_t)wrow * kDd + k0 + 16 * h;
      bf16x16 bh = *(const bf16x16*)(Wh + woff);
      bf16x16 bl = *(const bf16x16*)(Wl + woff);
      acc[j] = WMMA_BF16(ah, bh, acc[j]);
      acc[j] = WMMA_BF16(ah, bl, acc[j]);
      acc[j] = WMMA_BF16(al, bh, acc[j]);
    }
  }

  // C/D layout: VGPR r -> M = r + 8*half, N = lane&15
  #pragma unroll
  for (int j = 0; j < 5; ++j) {
    const int ntb = (ngrp * 5 + j) * 16;
    #pragma unroll
    for (int r = 0; r < 8; ++r) {
      const size_t row = (size_t)mtile * 16 + r + 8 * h;
      const int col = ntb + m;
      if (ntb < kN) Bu[row * kN + col]          = acc[j][r];
      else          Dy[row * kDa + (col - kN)]  = acc[j][r];
    }
  }
}

// ---------------- Phase 2: chunk-local scans --------------------------------

__global__ __launch_bounds__(64) void ssm_scan(const float* __restrict__ Bu,
                                               const float* __restrict__ lnl,
                                               const float* __restrict__ logdelta,
                                               float* __restrict__ xs,
                                               float* __restrict__ state) {
  const int c = blockIdx.x;              // chunk 0..15
  const int b = blockIdx.y;              // batch 0..7
  const int n = threadIdx.x;             // state 0..63
  const float a = expf(-expf(lnl[n]) * expf(logdelta[0]));
  const size_t base = ((size_t)b * kS + (size_t)c * kL) * kN + n;
  float x = 0.0f;
  #pragma unroll 4
  for (int t = 0; t < kL; ++t) {
    x = fmaf(x, a, Bu[base + (size_t)t * kN]);
    xs[base + (size_t)t * kN] = x;
  }
  state[(b * kNC + c) * kN + n] = x;     // chunk-final local state
}

// ---------------- Phase 3: sequential carry (tiny) --------------------------

__global__ __launch_bounds__(512) void ssm_carry(const float* __restrict__ state,
                                                 const float* __restrict__ lnl,
                                                 const float* __restrict__ logdelta,
                                                 float* __restrict__ carry) {
  const int tid = threadIdx.x;           // 512 = 8 batches * 64 states
  const int b = tid >> 6, n = tid & 63;
  const float aL = expf(-expf(lnl[n]) * expf(logdelta[0]) * (float)kL);
  float x = 0.0f;                        // x_in of chunk 0
  for (int c = 0; c < kNC; ++c) {
    carry[(b * kNC + c) * kN + n] = x;
    x = fmaf(x, aL, state[(b * kNC + c) * kN + n]);
  }
}

// ---------------- Phase 4: output GEMM with fused carry fixup ---------------
// y(16x16 tile) = (xs_local + a^(t+1)*carry)(16x64) @ C^T + Dy

__global__ __launch_bounds__(32) void ssm_gemm_out(const float* __restrict__ xs,
                                                   const float* __restrict__ carry,
                                                   const float* __restrict__ powt,
                                                   const __bf16* __restrict__ Ch,
                                                   const __bf16* __restrict__ Cl,
                                                   const float* __restrict__ Dy,
                                                   float* __restrict__ y) {
  const int mtile = blockIdx.x;          // 0..1023
  const int nt    = blockIdx.y;          // 0..15 (16 cols each of 256)
  const int l = threadIdx.x;
  const int h = l >> 4;
  const int m = l & 15;
  const size_t row0 = (size_t)mtile * 16;
  const int s0 = (int)(row0 % kS);
  const int b  = (int)(row0 / kS);
  const int c  = s0 / kL;                // constant across the 16-row tile
  const int t  = (s0 + m) % kL;          // per-lane position in chunk

  f32x8 acc;
  #pragma unroll
  for (int r = 0; r < 8; ++r)
    acc[r] = Dy[(row0 + r + 8 * h) * kDa + nt * 16 + m];

  const float* carr = carry + ((size_t)b * kNC + c) * kN;
  const float* pw   = powt + (size_t)t * kN;
  const size_t rowA = row0 + m;

  #pragma unroll
  for (int k0 = 0; k0 < kN; k0 += 32) {
    const size_t base = rowA * kN + k0;
    f32x4 av[4];
    av[0] = *(const f32x4*)(xs + base + 16 * h);
    av[1] = *(const f32x4*)(xs + base + 16 * h + 4);
    av[2] = *(const f32x4*)(xs + base + 16 * h + 8);
    av[3] = *(const f32x4*)(xs + base + 16 * h + 12);

    bf16x16 ah, al;
    #pragma unroll
    for (int i = 0; i < 16; ++i) {
      const int nidx = k0 + 16 * h + i;
      const float xv = av[i >> 2][i & 3] + pw[nidx] * carr[nidx];  // carry fixup
      __bf16 hi, lo;
      bsplit(xv, hi, lo);
      ah[i] = hi;
      al[i] = lo;
    }

    const size_t coff = (size_t)(nt * 16 + m) * kN + k0 + 16 * h;
    bf16x16 bh = *(const bf16x16*)(Ch + coff);
    bf16x16 bl = *(const bf16x16*)(Cl + coff);
    acc = WMMA_BF16(ah, bh, acc);
    acc = WMMA_BF16(ah, bl, acc);
    acc = WMMA_BF16(al, bh, acc);
  }

  #pragma unroll
  for (int r = 0; r < 8; ++r)
    y[(row0 + r + 8 * h) * kDa + nt * 16 + m] = acc[r];
}

// ---------------- Launch ----------------------------------------------------

extern "C" void kernel_launch(void* const* d_in, const int* in_sizes, int n_in,
                              void* d_out, int out_size, void* d_ws, size_t ws_size,
                              hipStream_t stream) {
  const float* u        = (const float*)d_in[0];
  const float* uref     = (const float*)d_in[1];
  const float* lnl      = (const float*)d_in[2];
  const float* Be       = (const float*)d_in[3];
  const float* Bf       = (const float*)d_in[4];
  const float* Cmat     = (const float*)d_in[5];
  const float* De       = (const float*)d_in[6];
  const float* Df       = (const float*)d_in[7];
  const float* logdelta = (const float*)d_in[8];
  float* y = (float*)d_out;

  // workspace layout (bytes)
  constexpr size_t SZ_W   = (size_t)kNW * kDd * 2;      // 640 KB
  constexpr size_t SZ_C   = (size_t)kDa * kN * 2;       // 32 KB
  constexpr size_t SZ_POW = (size_t)kL * kN * 4;        // 32 KB
  constexpr size_t SZ_BU  = (size_t)kM * kN * 4;        // 4 MB
  constexpr size_t SZ_ST  = (size_t)kB * kNC * kN * 4;  // 32 KB
  constexpr size_t OFF_WH  = 0;
  constexpr size_t OFF_WL  = OFF_WH + SZ_W;
  constexpr size_t OFF_CH  = OFF_WL + SZ_W;
  constexpr size_t OFF_CL  = OFF_CH + SZ_C;
  constexpr size_t OFF_POW = OFF_CL + SZ_C;
  constexpr size_t OFF_BU  = OFF_POW + SZ_POW;
  constexpr size_t OFF_XS  = OFF_BU + SZ_BU;
  constexpr size_t OFF_ST  = OFF_XS + SZ_BU;
  constexpr size_t OFF_CA  = OFF_ST + SZ_ST;
  constexpr size_t OFF_DY  = OFF_CA + SZ_ST;

  char* ws = (char*)d_ws;
  __bf16* Wh   = (__bf16*)(ws + OFF_WH);
  __bf16* Wl   = (__bf16*)(ws + OFF_WL);
  __bf16* Ch   = (__bf16*)(ws + OFF_CH);
  __bf16* Cl   = (__bf16*)(ws + OFF_CL);
  float*  powt = (float*)(ws + OFF_POW);
  float*  Bu   = (float*)(ws + OFF_BU);
  float*  xs   = (float*)(ws + OFF_XS);
  float*  st   = (float*)(ws + OFF_ST);
  float*  ca   = (float*)(ws + OFF_CA);
  float*  Dy   = (float*)(ws + OFF_DY);

  ssm_prep_w<<<(kNW * kDd) / 256, 256, 0, stream>>>(lnl, logdelta, Be, Bf, De, Df, Wh, Wl);
  ssm_prep_c<<<(kDa * kN) / 256, 256, 0, stream>>>(Cmat, Ch, Cl);
  ssm_prep_pow<<<(kL * kN) / 256, 256, 0, stream>>>(lnl, logdelta, powt);
  ssm_gemm_in<<<dim3(kM / 16, 4), 32, 0, stream>>>(u, uref, Wh, Wl, Bu, Dy);
  ssm_scan<<<dim3(kNC, kB), 64, 0, stream>>>(Bu, lnl, logdelta, xs, st);
  ssm_carry<<<1, 512, 0, stream>>>(st, lnl, logdelta, ca);
  ssm_gemm_out<<<dim3(kM / 16, kDa / 16), 32, 0, stream>>>(xs, ca, powt, Ch, Cl, Dy, y);
}
